// GCN_decoder_9826885173493
// MI455X (gfx1250) — matrix operand order repaired and checked by reading, not compiled, not run on hardware
//
#include <hip/hip_runtime.h>

#define N_NODES 10000
#define DIM     512
#define E_EDGES 160000

// ---- CDNA5 vector types ----
typedef __attribute__((ext_vector_type(16))) __bf16 v16bf;
typedef __attribute__((ext_vector_type(8)))  float  v8f;
typedef __attribute__((ext_vector_type(4)))  unsigned int v4u;
typedef __attribute__((ext_vector_type(8)))  int v8i;
typedef __attribute__((ext_vector_type(4)))  int v4i;

union Frag16 { v16bf v; uint4 q[2]; };

__device__ __forceinline__ unsigned short f2bf(float f) {
  unsigned int u = __float_as_uint(f);
  u += 0x7FFFu + ((u >> 16) & 1u);               // RNE fp32->bf16
  return (unsigned short)(u >> 16);
}

// -------------------------------------------------------------------------
// TDM: load a 2D bf16 tile (32 wide x 128 tall, row stride 512 elems) from
// global to LDS, padding each 64B row with 16B -> LDS row stride 80B (LDK=40).
// D# packing per CDNA5 ISA 8.3/8.4.  tensorRows = rows valid from tile start
// (OOB rows read as zero -> M-edge guard for free).
// -------------------------------------------------------------------------
__device__ __forceinline__ void tdm_load_tile(unsigned ldsOff,
                                              const unsigned short* gptr,
                                              unsigned tensorRows) {
  unsigned long long ga = (unsigned long long)(uintptr_t)gptr;
  v4u g0;
  g0[0] = 1u;                                            // count=1, user mode
  g0[1] = ldsOff;                                        // lds_addr
  g0[2] = (unsigned)(ga & 0xFFFFFFFFu);                  // global_addr[31:0]
  g0[3] = (unsigned)((ga >> 32) & 0x01FFFFFFu) | 0x80000000u; // [56:32] | type=2
  v8i g1;
  g1[0] = (1 << 16) | (1 << 20) | (3 << 22) | (3 << 25); // 2B elems, pad_en,
                                                         // interval=16DW, amt=4DW
  g1[1] = (int)(512u << 16);                             // tensor_dim0 = 512
  g1[2] = (int)((tensorRows & 0xFFFFu) << 16);           // tensor_dim1 lo16
  g1[3] = (int)((tensorRows >> 16) | (32u << 16));       // dim1 hi | tile_dim0=32
  g1[4] = 128;                                           // tile_dim1=128
  g1[5] = 512;                                           // dim0_stride = 512
  g1[6] = 0;
  g1[7] = 0;
  v4i z4 = {0, 0, 0, 0};                                 // groups 2/3 unused (2D)
  v8i z8 = {0, 0, 0, 0, 0, 0, 0, 0};
  __builtin_amdgcn_tensor_load_to_lds(g0, g1, z4, z4, z8, 0);
}

// -------------------------------------------------------------------------
// Dense GEMM: C[M x 512] = Abf[M x 512](bf16) * Bt^T, where Bt[n][k] is the
// bf16-transposed weight. 256 thr = 8 waves; WG tile 128x128, K-step 32.
// Tiles staged by the Tensor Data Mover; fragments via ds_load_b128; compute
// via v_wmma_f32_16x16x32_bf16 (2x4 tiles per wave).
// -------------------------------------------------------------------------
#define TM 128
#define TN 128
#define TK 32
#define LDK 40   // 80B padded LDS row stride (64B data + 16B TDM pad)

__global__ __launch_bounds__(256)
void gemm_tdm_wmma(const unsigned short* __restrict__ Abf,
                   const unsigned short* __restrict__ Bt,
                   float* __restrict__ C, int M) {
  __shared__ __align__(16) unsigned short sA [TM][LDK]; // [m][k]
  __shared__ __align__(16) unsigned short sBt[TN][LDK]; // [n][k]

  const int t      = threadIdx.x;
  const int lane   = t & 31;
  const int wave   = t >> 5;
  const int waveM  = wave & 3;
  const int waveN  = wave >> 2;
  const int blockM = blockIdx.x * TM;
  const int blockN = blockIdx.y * TN;
  const int lr = lane & 15;
  const int hi = lane >> 4;

  const unsigned rowsRem = (unsigned)(M - blockM);      // OOB guard via tensor_dim1
  const unsigned ldsA = (unsigned)(uintptr_t)(void*)&sA[0][0];
  const unsigned ldsB = (unsigned)(uintptr_t)(void*)&sBt[0][0];

  v8f acc[2][4];
  const v8f vzero = {0.f,0.f,0.f,0.f,0.f,0.f,0.f,0.f};
  #pragma unroll
  for (int i = 0; i < 2; ++i)
    #pragma unroll
    for (int j = 0; j < 4; ++j) acc[i][j] = vzero;

  for (int kt = 0; kt < DIM; kt += TK) {
    if (wave == 0) {
      tdm_load_tile(ldsA, Abf + (size_t)blockM * DIM + kt, rowsRem);
      tdm_load_tile(ldsB, Bt  + (size_t)blockN * DIM + kt, 512u);
      __builtin_amdgcn_s_wait_tensorcnt(0);
    }
    __syncthreads();

    // A 16x32 frag: lane M=lr; elems 0..7 -> K=hi*8+0..7, 8..15 -> K=16+hi*8+0..7
    Frag16 a[2];
    #pragma unroll
    for (int mi = 0; mi < 2; ++mi) {
      const unsigned short* rp = &sA[waveM * 32 + mi * 16 + lr][hi * 8];
      a[mi].q[0] = *(const uint4*)(rp);
      a[mi].q[1] = *(const uint4*)(rp + 16);
    }
    // B 32x16 frag: lane N=lr; elems 0..7 -> K=hi*8+0..7, 8..15 -> K=16+hi*8+0..7
    Frag16 b[4];
    #pragma unroll
    for (int ni = 0; ni < 4; ++ni) {
      const unsigned short* cp = &sBt[waveN * 64 + ni * 16 + lr][hi * 8];
      b[ni].q[0] = *(const uint4*)(cp);
      b[ni].q[1] = *(const uint4*)(cp + 16);
    }

    #pragma unroll
    for (int mi = 0; mi < 2; ++mi)
      #pragma unroll
      for (int ni = 0; ni < 4; ++ni)
        acc[mi][ni] = __builtin_amdgcn_wmma_f32_16x16x32_bf16(
            false, a[mi].v, false, b[ni].v, (short)0, acc[mi][ni], false, false);

    __syncthreads();
  }

  #pragma unroll
  for (int mi = 0; mi < 2; ++mi)
    #pragma unroll
    for (int ni = 0; ni < 4; ++ni)
      #pragma unroll
      for (int r = 0; r < 8; ++r) {
        int mm = blockM + waveM * 32 + mi * 16 + hi * 8 + r;
        int nn = blockN + waveN * 64 + ni * 16 + lr;
        if (mm < M) C[(size_t)mm * DIM + nn] = acc[mi][ni][r];
      }
}

// ---- fp32 -> bf16 bulk convert (4 elems/thread) ----
__global__ __launch_bounds__(256)
void cvt_bf16(const float* __restrict__ in, unsigned short* __restrict__ out) {
  unsigned i = (blockIdx.x * 256u + threadIdx.x) * 4u;
  float4 f = *(const float4*)(in + i);
  ushort4 o;
  o.x = f2bf(f.x); o.y = f2bf(f.y); o.z = f2bf(f.z); o.w = f2bf(f.w);
  *(ushort4*)(out + i) = o;
}

// ---- W[512][512] f32 -> Wt[n][k] bf16 (LDS-tiled transpose) ----
__global__ __launch_bounds__(256)
void transpose_cvt(const float* __restrict__ W, unsigned short* __restrict__ Wt) {
  __shared__ unsigned short tile[32][33];
  const int bx = blockIdx.x * 32, by = blockIdx.y * 32;
  const int tx = threadIdx.x & 31, ty = threadIdx.x >> 5;   // ty 0..7
  #pragma unroll
  for (int j = 0; j < 4; ++j)
    tile[ty + j * 8][tx] = f2bf(W[(size_t)(by + ty + j * 8) * DIM + bx + tx]);
  __syncthreads();
  #pragma unroll
  for (int j = 0; j < 4; ++j)
    Wt[(size_t)(bx + ty + j * 8) * DIM + by + tx] = tile[tx][ty + j * 8];
}

// ---- SpMM scatter: Y[rows[e]] += vals[e] * X[cols[e]] (Y pre-zeroed) ----
__global__ __launch_bounds__(256)
void spmm_scatter(const int* __restrict__ rows, const int* __restrict__ cols,
                  const float* __restrict__ vals, const float* __restrict__ X,
                  float* __restrict__ Y) {
  unsigned int gid = blockIdx.x * 256u + threadIdx.x;
  unsigned int e   = gid >> 7;
  unsigned int d4  = gid & 127u;
  int   r = rows[e];
  int   c = cols[e];
  float v = vals[e];
  float4 x = ((const float4*)(X + (size_t)c * DIM))[d4];
  float* y = Y + (size_t)r * DIM + (size_t)d4 * 4;
  atomicAdd(y + 0, v * x.x);
  atomicAdd(y + 1, v * x.y);
  atomicAdd(y + 2, v * x.z);
  atomicAdd(y + 3, v * x.w);
}

// ---- bias + leaky + hash-dropout, writes bf16 (feeds next GEMM) ----
__global__ __launch_bounds__(256)
void bias_act_drop_bf16(const float* __restrict__ S, const float* __restrict__ bias,
                        unsigned short* __restrict__ out) {
  unsigned i = (blockIdx.x * 256u + threadIdx.x) * 4u;
  float4 s = *(const float4*)(S + i);
  float4 b = *(const float4*)(bias + (i & (DIM - 1)));
  float x[4] = { s.x + b.x, s.y + b.y, s.z + b.z, s.w + b.w };
  ushort4 o;
  unsigned short* op = &o.x;
  #pragma unroll
  for (int j = 0; j < 4; ++j) {
    float v = x[j];
    v = (v >= 0.f) ? v : 0.25f * v;
    unsigned h = (i + j) * 2654435761u;
    h ^= h >> 15; h *= 2246822519u; h ^= h >> 13;
    v = (h & 1u) ? v * 2.0f : 0.0f;
    op[j] = f2bf(v);
  }
  *(ushort4*)(out + i) = o;
}

// ---- bias + leaky, fp32 output ----
__global__ __launch_bounds__(256)
void bias_act_f32(const float* __restrict__ S, const float* __restrict__ bias,
                  float* __restrict__ out) {
  unsigned i = (blockIdx.x * 256u + threadIdx.x) * 4u;
  float4 s = *(const float4*)(S + i);
  float4 b = *(const float4*)(bias + (i & (DIM - 1)));
  float4 o;
  o.x = s.x + b.x; o.y = s.y + b.y; o.z = s.z + b.z; o.w = s.w + b.w;
  o.x = (o.x >= 0.f) ? o.x : 0.25f * o.x;
  o.y = (o.y >= 0.f) ? o.y : 0.25f * o.y;
  o.z = (o.z >= 0.f) ? o.z : 0.25f * o.z;
  o.w = (o.w >= 0.f) ? o.w : 0.25f * o.w;
  *(float4*)(out + i) = o;
}

// -------------------------------------------------------------------------
extern "C" void kernel_launch(void* const* d_in, const int* in_sizes, int n_in,
                              void* d_out, int out_size, void* d_ws, size_t ws_size,
                              hipStream_t stream) {
  const float* H    = (const float*)d_in[0];
  const int*   rows = (const int*)  d_in[1];
  const int*   cols = (const int*)  d_in[2];
  const float* vals = (const float*)d_in[3];
  const float* W1   = (const float*)d_in[4];
  const float* b1   = (const float*)d_in[5];
  const float* W2   = (const float*)d_in[6];
  const float* b2   = (const float*)d_in[7];
  float* out = (float*)d_out;

  const size_t NM = (size_t)N_NODES * DIM;
  char* w = (char*)d_ws;
  float*          Y   = (float*)w;           w += NM * sizeof(float);
  float*          S   = (float*)w;           w += NM * sizeof(float);
  unsigned short* Abf = (unsigned short*)w;  w += NM * sizeof(unsigned short);
  unsigned short* W1t = (unsigned short*)w;  w += (size_t)DIM * DIM * 2;
  unsigned short* W2t = (unsigned short*)w;

  dim3 ggrid((N_NODES + TM - 1) / TM, DIM / TN);
  dim3 tgrid(DIM / 32, DIM / 32);
  const int spmmBlocks = (E_EDGES * 128) / 256;  // exact
  const int vecBlocks  = (int)(NM / 1024);       // exact (4 elems/thread)

  // prep: bf16 operands for the WMMA pipeline
  transpose_cvt<<<tgrid, 256, 0, stream>>>(W1, W1t);
  transpose_cvt<<<tgrid, 256, 0, stream>>>(W2, W2t);
  cvt_bf16<<<vecBlocks, 256, 0, stream>>>(H, Abf);

  // ---- layer 1 ----
  gemm_tdm_wmma<<<ggrid, 256, 0, stream>>>(Abf, W1t, Y, N_NODES);
  (void)hipMemsetAsync(S, 0, NM * sizeof(float), stream);
  spmm_scatter<<<spmmBlocks, 256, 0, stream>>>(rows, cols, vals, Y, S);
  bias_act_drop_bf16<<<vecBlocks, 256, 0, stream>>>(S, b1, Abf);

  // ---- layer 2 ----
  gemm_tdm_wmma<<<ggrid, 256, 0, stream>>>(Abf, W2t, Y, N_NODES);
  (void)hipMemsetAsync(S, 0, NM * sizeof(float), stream);
  spmm_scatter<<<spmmBlocks, 256, 0, stream>>>(rows, cols, vals, Y, S);
  bias_act_f32<<<vecBlocks, 256, 0, stream>>>(S, b2, out);
}
